// RecurrentNetwork_19628000543215
// MI455X (gfx1250) — matrix operand order repaired and checked by reading, not compiled, not run on hardware
//
#include <hip/hip_runtime.h>
#include <hip/hip_bf16.h>
#include <math.h>

typedef __attribute__((ext_vector_type(16))) _Float16 v16h;
typedef __attribute__((ext_vector_type(8)))  _Float16 v8h;
typedef __attribute__((ext_vector_type(8)))  float    v8f;

#define T_STEPS 1024
#define BATCH   64
#define NIN     32
#define NH      256
#define NOUT    16
#define ALPHA   0.1f
#define RSTRIDE 264   // 256 + 8 halfs pad -> +4 LDS banks per row, conflict-free b128 A loads

__device__ __forceinline__ v8f wmma_f16(v16h a, v16h b, v8f c) {
  // 8 args: (neg_a, A, neg_b, B, c_mod, C, reuse_a, reuse_b)
  return __builtin_amdgcn_wmma_f32_16x16x32_f16(false, a, false, b, (short)0, c, false, false);
}

__device__ __forceinline__ float fast_tanh(float v) {
#if __has_builtin(__builtin_amdgcn_tanhf)
  return __builtin_amdgcn_tanhf(v);          // v_tanh_f32 (CDNA5 hardware transcendental)
#elif __has_builtin(__builtin_amdgcn_tanh_f32)
  return __builtin_amdgcn_tanh_f32(v);
#else
  return tanhf(v);
#endif
}

// One recurrence step. PIN = LDS ping-pong buffer holding r_{t-1}.
// QCUR/UCUR were prefetched a full iteration earlier (HBM latency hidden by
// the previous step's wmma chain); this body issues the prefetch for t+1.
#define STEP_BODY(T_IDX, PIN, QCUR, UCUR, QNXT, UNXT)                         \
  {                                                                           \
    const int t  = (T_IDX);                                                   \
    const int tp = (t + 1 < T_STEPS) ? (t + 1) : t;  /* clamped prefetch */   \
    {                                                                         \
      const float* qn_p = noise + ((size_t)tp*BATCH + b0)*NH;                 \
      _Pragma("unroll")                                                       \
      for (int j = 0; j < 8; ++j)                                             \
        QNXT[j] = qn_p[(half*8 + j)*NH + n];                                  \
      const float* un_p = inputs + ((size_t)tp*BATCH + b0 + lm)*NIN;          \
      _Pragma("unroll")                                                       \
      for (int i = 0; i < 8; ++i) {                                           \
        UNXT[i]   = un_p[half*8 + i];                                         \
        UNXT[8+i] = un_p[half*8 + 16 + i];                                    \
      }                                                                       \
    }                                                                         \
    /* A = r_{t-1}[16x256] from LDS buffer PIN; Wrec wmma chain */            \
    v8f acc = {0.f,0.f,0.f,0.f,0.f,0.f,0.f,0.f};                              \
    _Pragma("unroll")                                                         \
    for (int kb = 0; kb < 8; ++kb) {                                          \
      const int k0 = kb*32 + half*8;                                          \
      v8h lo = *(const v8h*)&rbuf[PIN][lm][k0];                               \
      v8h hi = *(const v8h*)&rbuf[PIN][lm][k0 + 16];                          \
      v16h a;                                                                 \
      _Pragma("unroll")                                                       \
      for (int i = 0; i < 8; ++i) { a[i] = lo[i]; a[8+i] = hi[i]; }           \
      acc = wmma_f16(a, bw[kb], acc);                                         \
    }                                                                         \
    /* + u_t @ Win from registers prefetched last iteration */                \
    {                                                                         \
      v16h au;                                                                \
      _Pragma("unroll")                                                       \
      for (int i = 0; i < 16; ++i) au[i] = (_Float16)UCUR[i];                 \
      acc = wmma_f16(au, bu, acc);                                            \
    }                                                                         \
    /* leaky update with prefetched q; stream states; write r_t */            \
    {                                                                         \
      float* xs = out_x + ((size_t)t*BATCH + b0)*NH;                          \
      _Pragma("unroll")                                                       \
      for (int j = 0; j < 8; ++j) {                                           \
        const int m = half*8 + j;                                             \
        float xn = (1.0f - ALPHA)*x[j] + ALPHA*(acc[j] + brec_n) + QCUR[j];   \
        x[j] = xn;                                                            \
        xs[m*NH + n] = xn;                                                    \
        rbuf[(PIN) ^ 1][m][n] = (_Float16)fast_tanh(xn);                      \
      }                                                                       \
    }                                                                         \
    __syncthreads();                                                          \
  }

// ---------------------------------------------------------------------------
// Kernel 1: the serial recurrence. Uniform, branch-free per-step body:
//   16 ds_load_b128 (A), 8+1 v_wmma, v_tanh_f32 update, streaming stores,
//   one barrier; q_t/u_t HBM loads software-pipelined one full step ahead.
// ---------------------------------------------------------------------------
__global__ __launch_bounds__(512, 1) void rnn_recur_kernel(
    const float* __restrict__ inputs, const float* __restrict__ noise,
    const float* __restrict__ x0,     const float* __restrict__ Win,
    const float* __restrict__ Wrec,   const float* __restrict__ brec,
    float* __restrict__ out_x)
{
  // double-buffered r = tanh(x) tile: [buf][row 0..15][K 0..255], f16
  __shared__ _Float16 rbuf[2][16][RSTRIDE];

  const int lane = threadIdx.x & 31;
  const int wave = threadIdx.x >> 5;     // 0..15 : column tile of N
  const int half = lane >> 4;            // lane group (ISA A/B K-offset select)
  const int lm   = lane & 15;
  const int b0   = blockIdx.x * 16;      // batch-row slice for this WGP
  const int n    = wave * 16 + lm;       // column owned by this lane in B/C/D

  // ---- stationary B operands (VGPR-resident f16 for all 1024 steps) ----
  // B layout (16-bit, KxN=32x16): lane lm = N; half i -> K = 8*half + i + (i & 8)
  v16h bw[8];                            // Wrec [256,256] row-major, 8 K-chunks of 32
  #pragma unroll
  for (int kb = 0; kb < 8; ++kb) {
    #pragma unroll
    for (int i = 0; i < 16; ++i)
      bw[kb][i] = (_Float16)Wrec[(kb*32 + half*8 + i + (i & 8))*NH + n];
  }
  v16h bu;                               // Win [32,256], single K=32 chunk
  #pragma unroll
  for (int i = 0; i < 16; ++i)
    bu[i] = (_Float16)Win[(half*8 + i + (i & 8))*NH + n];

  const float brec_n = brec[n];

  // ---- recurrent state x in WMMA C/D layout: VGPR j <-> row m = half*8 + j ----
  float x[8];
  #pragma unroll
  for (int j = 0; j < 8; ++j)
    x[j] = x0[(size_t)(b0 + half*8 + j)*NH + n];

  #pragma unroll
  for (int j = 0; j < 8; ++j)
    rbuf[0][half*8 + j][n] = (_Float16)fast_tanh(x[j]);

  // ---- prologue prefetch: q_0 and u_0 ----
  float qA[8], qB[8], uA[16], uB[16];
  {
    const float* q0 = noise + (size_t)b0*NH;
    #pragma unroll
    for (int j = 0; j < 8; ++j)
      qA[j] = q0[(half*8 + j)*NH + n];
    const float* u0 = inputs + (size_t)(b0 + lm)*NIN;
    #pragma unroll
    for (int i = 0; i < 8; ++i) {
      uA[i]   = u0[half*8 + i];
      uA[8+i] = u0[half*8 + 16 + i];
    }
  }
  __syncthreads();

  // time loop unrolled x2: buffer rotation is pure register renaming,
  // and the LDS ping-pong index is a compile-time constant per body.
  for (int t = 0; t < T_STEPS; t += 2) {
    STEP_BODY(t,     0, qA, uA, qB, uB)
    STEP_BODY(t + 1, 1, qB, uB, qA, uA)
  }
}

// ---------------------------------------------------------------------------
// Kernel 2: z = tanh(states) @ Wout + bout, batched over all T*B rows.
// Pure parallel GEMM [65536,256]x[256,16]; states likely L2-resident (64MB
// vs 192MB L2). One 16-row tile per wave, 8 waves per block, 512 blocks.
// ---------------------------------------------------------------------------
__global__ __launch_bounds__(256, 1) void rnn_zout_kernel(
    const float* __restrict__ states, const float* __restrict__ Wout,
    const float* __restrict__ bout,   float* __restrict__ out_z)
{
  // Wout B-tiles pre-converted to f16 in per-lane ISA layout: [kb][lane][i]
  __shared__ _Float16 wout_lds[8][32][16];

  const int lane = threadIdx.x & 31;
  const int wave = threadIdx.x >> 5;   // 0..7
  const int half = lane >> 4;
  const int lm   = lane & 15;

  if (wave == 0) {
    #pragma unroll
    for (int kb = 0; kb < 8; ++kb) {
      #pragma unroll
      for (int i = 0; i < 16; ++i)
        wout_lds[kb][lane][i] = (_Float16)Wout[(kb*32 + half*8 + i + (i & 8))*NOUT + lm];
    }
  }
  __syncthreads();

  const int    tile = blockIdx.x * 8 + wave;        // 0..4095 row-tiles of T*B
  const size_t row  = (size_t)tile*16 + lm;         // this lane's A row
  const float* xr   = states + row*NH;

  v8f acc = {0.f,0.f,0.f,0.f,0.f,0.f,0.f,0.f};
  #pragma unroll
  for (int kb = 0; kb < 8; ++kb) {
    const int k0 = kb*32 + half*8;
    v16h a;
    #pragma unroll
    for (int i = 0; i < 8; ++i) {
      a[i]   = (_Float16)fast_tanh(xr[k0 + i]);
      a[8+i] = (_Float16)fast_tanh(xr[k0 + 16 + i]);
    }
    v16h bo = *(const v16h*)&wout_lds[kb][lane][0];
    acc = wmma_f16(a, bo, acc);
  }

  const float bout_n = bout[lm];
  float* z = out_z + (size_t)tile * 16 * NOUT;
  #pragma unroll
  for (int j = 0; j < 8; ++j)
    z[(half*8 + j)*NOUT + lm] = acc[j] + bout_n;
}

extern "C" void kernel_launch(void* const* d_in, const int* in_sizes, int n_in,
                              void* d_out, int out_size, void* d_ws, size_t ws_size,
                              hipStream_t stream) {
  const float* inputs = (const float*)d_in[0];   // [T,B,NIN]
  const float* noise  = (const float*)d_in[1];   // [T,B,N]
  const float* x0     = (const float*)d_in[2];   // [B,N]
  const float* Win    = (const float*)d_in[3];   // [NIN,N]
  const float* Wrec   = (const float*)d_in[4];   // [N,N]
  const float* brec   = (const float*)d_in[5];   // [N]
  const float* Wout   = (const float*)d_in[6];   // [N,NOUT]
  const float* bout   = (const float*)d_in[7];   // [NOUT]

  float* out_z = (float*)d_out;                                  // [T,B,NOUT]
  float* out_x = out_z + (size_t)T_STEPS * BATCH * NOUT;         // [T,B,N]

  rnn_recur_kernel<<<BATCH/16, 512, 0, stream>>>(
      inputs, noise, x0, Win, Wrec, brec, out_x);

  rnn_zout_kernel<<<(T_STEPS*BATCH)/(16*8), 256, 0, stream>>>(
      out_x, Wout, bout, out_z);
}
